// EdgeUpdateNet_9131100471463
// MI455X (gfx1250) — matrix operand order repaired and checked by reading, not compiled, not run on hardware
//
#include <hip/hip_runtime.h>

// ---------------------------------------------------------------------------
// EdgeUpdateNet for MI455X (gfx1250, wave32, WMMA bf16)
//
// Algebraic refactor: cat([h[src],h[dst],ea]) @ w0 is split into node-level
// GEMMs hs = h@w0[0:H], hd = h@w0[H:2H] plus edge-level ea@w0[2H:].
// All GEMMs run through one tiled v_wmma_f32_16x16x32_bf16 kernel
// (fp32 accum, K-step 64, double-buffered LDS, register-staged pipeline).
// ---------------------------------------------------------------------------

#define C_    64
#define NG_   128
#define H_    1088
#define OFM_  1024

typedef __attribute__((ext_vector_type(16))) __bf16 v16bf;
typedef __attribute__((ext_vector_type(8)))  __bf16 v8bf;
typedef __attribute__((ext_vector_type(4)))  __bf16 v4bf;
typedef __attribute__((ext_vector_type(8)))  float  v8f;

__device__ __forceinline__ __bf16 f2bf(float f) {
  unsigned u = __builtin_bit_cast(unsigned, f);
  unsigned r = u + 0x7FFFu + ((u >> 16) & 1u);      // round-to-nearest-even
  unsigned short s = (unsigned short)(r >> 16);
  return __builtin_bit_cast(__bf16, s);
}

__device__ __forceinline__ float sspf(float x) {    // softplus(x) - log(2)
  float sp = (x > 20.0f) ? x : log1pf(expf(x));
  return sp - 0.69314718055994530942f;
}

#define FLAG_SSP 1
#define FLAG_ACC 2

// ---------------------------------------------------------------------------
// Generic tiled GEMM: Y[M,Nn] (= or +=) act(X[M,K] @ W[K,Nn] + bias)
// Requires K % 64 == 0 (true for all shapes here: 1088, 128, 64).
// Block = 128 threads (4 waves). Tile 64(M) x 64(N), K-step 64.
// Each wave: 16 rows x 64 cols = 4 f32 accumulators, 8 WMMAs per K-slab.
// Double-buffered LDS; next slab's global loads issue before current WMMAs.
// ---------------------------------------------------------------------------
__global__ __launch_bounds__(128) void wmma_gemm_kernel(
    const float* __restrict__ X, int ldx,
    const float* __restrict__ W, int ldw,
    const float* __restrict__ bias,
    float* __restrict__ Y, int ldy,
    int M, int Nn, int K, int flags)
{
  __shared__ __align__(16) __bf16 Xs[2][64][64];   // [buf][row][k]  16 KB
  __shared__ __align__(16) __bf16 WsT[2][64][64];  // [buf][col][k]  16 KB

  const int tid  = threadIdx.x;
  const int wave = tid >> 5;
  const int lane = tid & 31;
  const int gm0  = blockIdx.y * 64;
  const int gn0  = blockIdx.x * 64;
  const bool fullN = (gn0 + 64) <= Nn;

  v8f acc[4] = {};

  float4 xr[8], wr[8];

  // ---- register-stage a 64x64 X slab (coalesced float4) ----
  auto load_x = [&](int k0) {
    #pragma unroll
    for (int i = 0; i < 8; ++i) {
      int idx = tid + i * 128;
      int row = idx >> 4;            // 0..63
      int c4  = (idx & 15) * 4;      // 0,4,..,60
      int gr  = gm0 + row;
      float4 v = make_float4(0.f, 0.f, 0.f, 0.f);
      if (gr < M) {
        const float* p = X + (long)gr * ldx + k0 + c4;
        v = *(const float4*)p;
        if (k0 + 64 < K) __builtin_prefetch(p + 64, 0, 1);   // slab after next
      }
      xr[i] = v;
    }
  };
  // ---- register-stage a 64x64 W slab ----
  auto load_w = [&](int k0) {
    if (fullN) {
      #pragma unroll
      for (int i = 0; i < 8; ++i) {
        int idx = tid + i * 128;
        int kk  = idx >> 4;          // 0..63
        int c4  = (idx & 15) * 4;    // 0,4,..,60
        wr[i] = *(const float4*)(W + (long)(k0 + kk) * ldw + gn0 + c4);
      }
    } else {
      #pragma unroll
      for (int i = 0; i < 8; ++i) {
        int idx = tid + i * 128;
        int kk  = idx >> 4;
        int c4  = (idx & 15) * 4;
        float4 v = make_float4(0.f, 0.f, 0.f, 0.f);
        int col = gn0 + c4;
        if (col + 0 < Nn) v.x = W[(long)(k0 + kk) * ldw + col + 0];
        if (col + 1 < Nn) v.y = W[(long)(k0 + kk) * ldw + col + 1];
        if (col + 2 < Nn) v.z = W[(long)(k0 + kk) * ldw + col + 2];
        if (col + 3 < Nn) v.w = W[(long)(k0 + kk) * ldw + col + 3];
        wr[i] = v;
      }
    }
  };
  auto store_x = [&](int buf) {
    #pragma unroll
    for (int i = 0; i < 8; ++i) {
      int idx = tid + i * 128;
      int row = idx >> 4;
      int c4  = (idx & 15) * 4;
      v4bf pk;
      pk[0] = f2bf(xr[i].x); pk[1] = f2bf(xr[i].y);
      pk[2] = f2bf(xr[i].z); pk[3] = f2bf(xr[i].w);
      *(v4bf*)&Xs[buf][row][c4] = pk;
    }
  };
  auto store_w = [&](int buf) {      // transposed: WsT[col][k]
    #pragma unroll
    for (int i = 0; i < 8; ++i) {
      int idx = tid + i * 128;
      int kk  = idx >> 4;
      int c4  = (idx & 15) * 4;
      WsT[buf][c4 + 0][kk] = f2bf(wr[i].x);
      WsT[buf][c4 + 1][kk] = f2bf(wr[i].y);
      WsT[buf][c4 + 2][kk] = f2bf(wr[i].z);
      WsT[buf][c4 + 3][kk] = f2bf(wr[i].w);
    }
  };

  // ---- fragment loads + 8 WMMAs for one 64-wide K slab ----
  // A (16-bit 16x32 ISA layout): lane L holds row (L&15); lanes<16 hold
  // K{0..7,16..23}, lanes>=16 hold K{8..15,24..31} of the sub-slab.
  // B: lane L holds col (L&15); same K order as A (lanes<16 K{0..7,16..23},
  // lanes>=16 K{8..15,24..31}) from the transposed LDS image.
  auto compute = [&](int buf) {
    const int mrow = wave * 16 + (lane & 15);
    const int kb   = (lane >> 4) * 8;
    const int nl   = lane & 15;
    const int kb2  = (lane >> 4) * 16;
    union F { v16bf v; v8bf h[2]; };
    F A0, A1;
    A0.h[0] = *(const v8bf*)&Xs[buf][mrow][kb];
    A0.h[1] = *(const v8bf*)&Xs[buf][mrow][kb + 16];
    A1.h[0] = *(const v8bf*)&Xs[buf][mrow][32 + kb];
    A1.h[1] = *(const v8bf*)&Xs[buf][mrow][32 + kb + 16];
    #pragma unroll
    for (int nf = 0; nf < 4; ++nf) {
      const int n = nf * 16 + nl;
      F B0, B1;
      B0.h[0] = *(const v8bf*)&WsT[buf][n][kb];
      B0.h[1] = *(const v8bf*)&WsT[buf][n][kb + 16];
      B1.h[0] = *(const v8bf*)&WsT[buf][n][32 + kb];
      B1.h[1] = *(const v8bf*)&WsT[buf][n][32 + kb + 16];
      acc[nf] = __builtin_amdgcn_wmma_f32_16x16x32_bf16(false, A0.v, false, B0.v, (short)0, acc[nf], false, false);
      acc[nf] = __builtin_amdgcn_wmma_f32_16x16x32_bf16(false, A1.v, false, B1.v, (short)0, acc[nf], false, false);
    }
  };

  // ---- software pipeline: one barrier per K slab ----
  load_x(0); load_w(0);
  store_x(0); store_w(0);
  __syncthreads();

  const int nslabs = K >> 6;
  for (int s = 0; s < nslabs; ++s) {
    const int  cur  = s & 1;
    const bool more = (s + 1) < nslabs;
    if (more) { load_x((s + 1) << 6); load_w((s + 1) << 6); }  // HBM/L2 latency hides under WMMA
    compute(cur);
    if (more) { store_x(cur ^ 1); store_w(cur ^ 1); }
    __syncthreads();
  }

  // ---- epilogue: C/D layout — lane L: col=(L&15), rows v + 8*(L>>4) ----
  const int nl    = lane & 15;
  const int mbase = gm0 + wave * 16 + (lane >> 4) * 8;
  #pragma unroll
  for (int nf = 0; nf < 4; ++nf) {
    int gn = gn0 + nf * 16 + nl;
    if (gn < Nn) {
      float bv = bias ? bias[gn] : 0.f;
      #pragma unroll
      for (int v = 0; v < 8; ++v) {
        int gm = mbase + v;
        if (gm < M) {
          float val = acc[nf][v] + bv;
          if (flags & FLAG_SSP) val = sspf(val);
          long o = (long)gm * ldy + gn;
          if (flags & FLAG_ACC) Y[o] += val; else Y[o] = val;
        }
      }
    }
  }
}

// ---------------------------------------------------------------------------
// Elementwise / gather-scatter kernels
// ---------------------------------------------------------------------------
__global__ void build_h_kernel(const int* __restrict__ z, const float* __restrict__ emb,
                               const float* __restrict__ ofms, float* __restrict__ h, int N) {
  long idx = (long)blockIdx.x * blockDim.x + threadIdx.x;
  if (idx >= (long)N * H_) return;
  int n = (int)(idx / H_), j = (int)(idx % H_);
  h[idx] = (j < C_) ? emb[z[n] * C_ + j] : ofms[(long)n * OFM_ + (j - C_)];
}

__global__ void rbf_kernel(const float* __restrict__ d, float* __restrict__ ea, int E) {
  long idx = (long)blockIdx.x * blockDim.x + threadIdx.x;
  if (idx >= (long)E * NG_) return;
  int e = (int)(idx / NG_), g = (int)(idx % NG_);
  float ds  = d[e] * 0.1f;                       // d / cutoff
  float off = (float)g * (1.0f / 127.0f);        // linspace(0,1,128)
  float coeff = -0.5f * 127.0f * 127.0f;         // -0.5/delta^2
  float t = ds - off;
  float gv = expf(coeff * t * t);
  float ds2 = ds * ds;
  float ds5 = ds2 * ds2 * ds;
  float env = 1.0f + ds5 * (-21.0f + ds * (35.0f - 15.0f * ds));  // p=5 envelope
  env = (ds < 1.0f) ? env : 0.0f;
  ea[idx] = env * gv;
}

// pre = ssp(pre + hs[src] + hd[dst] + b0)   (width 2C = 128)
__global__ void edge_combine_kernel(const int* __restrict__ src, const int* __restrict__ dst,
                                    const float* __restrict__ hs, const float* __restrict__ hd,
                                    const float* __restrict__ b0, float* __restrict__ pre, int E) {
  long idx = (long)blockIdx.x * blockDim.x + threadIdx.x;
  if (idx >= (long)E * 128) return;
  int e = (int)(idx >> 7), j = (int)(idx & 127);
  float v = pre[idx] + hs[(long)src[e] * 128 + j] + hd[(long)dst[e] * 128 + j] + b0[j];
  pre[idx] = sspf(v);
}

// m[dst] += hq[src] * efeat   (segment_sum via f32 global atomics)
__global__ void scatter_kernel(const int* __restrict__ src, const int* __restrict__ dst,
                               const float* __restrict__ hq, const float* __restrict__ ef,
                               float* __restrict__ m, int E) {
  long idx = (long)blockIdx.x * blockDim.x + threadIdx.x;
  if (idx >= (long)E * 64) return;
  int e = (int)(idx >> 6), j = (int)(idx & 63);
  atomicAdd(&m[(long)dst[e] * 64 + j], hq[(long)src[e] * 64 + j] * ef[idx]);
}

__global__ void fill_kernel(float* __restrict__ p, long n, float v) {
  long i = (long)blockIdx.x * blockDim.x + threadIdx.x;
  if (i < n) p[i] = v;
}

// o[n] = dot(tn[n,:32], out1_w) + out1_b; out[batch[n]] += o[n]
__global__ void out_reduce_kernel(const float* __restrict__ tn, const float* __restrict__ w1,
                                  const float* __restrict__ b1, const int* __restrict__ batch,
                                  float* __restrict__ out, int N) {
  int n = blockIdx.x * blockDim.x + threadIdx.x;
  if (n >= N) return;
  float acc = b1[0];
  #pragma unroll
  for (int k = 0; k < 32; ++k) acc += tn[(long)n * 32 + k] * w1[k];
  atomicAdd(&out[batch[n]], acc);
}

// ---------------------------------------------------------------------------
// Host orchestration
// ---------------------------------------------------------------------------
static void gemm(hipStream_t s, const float* X, int ldx, const float* W, int ldw,
                 const float* bias, float* Y, int ldy, int M, int Nn, int K, int flags) {
  dim3 grid((Nn + 63) / 64, (M + 63) / 64);
  wmma_gemm_kernel<<<grid, 128, 0, s>>>(X, ldx, W, ldw, bias, Y, ldy, M, Nn, K, flags);
}

static inline long ew_blocks(long n) { return (n + 255) / 256; }

extern "C" void kernel_launch(void* const* d_in, const int* in_sizes, int n_in,
                              void* d_out, int out_size, void* d_ws, size_t ws_size,
                              hipStream_t stream) {
  const int N = in_sizes[0];
  const int E = in_sizes[2];

  int i = 0;
  const int*   z     = (const int*)d_in[i++];
  const int*   eidx  = (const int*)d_in[i++];
  const float* dist  = (const float*)d_in[i++];
  const float* ofms  = (const float*)d_in[i++];
  const int*   batch = (const int*)d_in[i++];
  const float* emb   = (const float*)d_in[i++];
  struct EU { const float *w0, *b0, *w1, *b1; } eu[3];
  for (int l = 0; l < 3; ++l) {
    eu[l].w0 = (const float*)d_in[i++]; eu[l].b0 = (const float*)d_in[i++];
    eu[l].w1 = (const float*)d_in[i++]; eu[l].b1 = (const float*)d_in[i++];
  }
  struct IT { const float *fc1_w,*fc1_b,*m0_w,*m0_b,*m1_w,*m1_b,*p0_w,*p0_b,*p1_w,*p1_b; } it[3];
  for (int l = 0; l < 3; ++l) {
    it[l].fc1_w=(const float*)d_in[i++]; it[l].fc1_b=(const float*)d_in[i++];
    it[l].m0_w =(const float*)d_in[i++]; it[l].m0_b =(const float*)d_in[i++];
    it[l].m1_w =(const float*)d_in[i++]; it[l].m1_b =(const float*)d_in[i++];
    it[l].p0_w =(const float*)d_in[i++]; it[l].p0_b =(const float*)d_in[i++];
    it[l].p1_w =(const float*)d_in[i++]; it[l].p1_b =(const float*)d_in[i++];
  }
  const float* out0_w = (const float*)d_in[i++];
  const float* out0_b = (const float*)d_in[i++];
  const float* out1_w = (const float*)d_in[i++];
  const float* out1_b = (const float*)d_in[i++];

  const int* src = eidx;        // edge_index[0]
  const int* dst = eidx + E;    // edge_index[1]

  // workspace carve-up (floats, 256B aligned chunks)
  float* ws = (float*)d_ws;
  size_t off = 0;
  auto carve = [&](size_t nf) { float* p = ws + off; off += (nf + 63) & ~(size_t)63; return p; };
  float* h    = carve((size_t)N * H_);       // node features [N,1088]
  float* ea   = carve((size_t)E * 128);      // edge_attr (<= [E,128])
  float* pre  = carve((size_t)E * 128);      // edge scratch [E,128] (e1/e2 halves reuse)
  float* hs   = carve((size_t)N * 128);
  float* hd   = carve((size_t)N * 128);
  float* hq   = carve((size_t)N * 64);
  float* m    = carve((size_t)N * 64);
  float* tn   = carve((size_t)N * 64);
  float* pre2 = pre + (size_t)E * 64;        // second half of pre
  (void)ws_size;

  // h = concat(emb[z], ofms);  ea = rbf(dist)
  build_h_kernel<<<ew_blocks((long)N * H_), 256, 0, stream>>>(z, emb, ofms, h, N);
  rbf_kernel<<<ew_blocks((long)E * NG_), 256, 0, stream>>>(dist, ea, E);

  int ea_w = NG_;  // current edge_attr width: 128 then 64
  for (int l = 0; l < 3; ++l) {
    const float* w0_src = eu[l].w0;
    const float* w0_dst = eu[l].w0 + (size_t)H_ * 128;
    const float* w0_e   = eu[l].w0 + (size_t)2 * H_ * 128;

    // node-level halves of the edge-update GEMM
    gemm(stream, h, H_, w0_src, 128, nullptr, hs, 128, N, 128, H_, 0);
    gemm(stream, h, H_, w0_dst, 128, nullptr, hd, 128, N, 128, H_, 0);
    // edge part: pre = ea @ w0_e
    gemm(stream, ea, ea_w, w0_e, 128, nullptr, pre, 128, E, 128, ea_w, 0);
    // pre = ssp(pre + hs[src] + hd[dst] + b0)
    edge_combine_kernel<<<ew_blocks((long)E * 128), 256, 0, stream>>>(src, dst, hs, hd, eu[l].b0, pre, E);
    // ea = pre @ w1 + b1            [E,64]
    gemm(stream, pre, 128, eu[l].w1, 64, eu[l].b1, ea, 64, E, 64, 128, 0);

    // InteractionBlock
    gemm(stream, h, H_, it[l].fc1_w, 64, it[l].fc1_b, hq, 64, N, 64, H_, 0);          // hq
    gemm(stream, ea, 64, it[l].m0_w, 64, it[l].m0_b, pre, 64, E, 64, 64, FLAG_SSP);   // e1
    gemm(stream, pre, 64, it[l].m1_w, 64, it[l].m1_b, pre2, 64, E, 64, 64, FLAG_SSP); // e2
    fill_kernel<<<ew_blocks((long)N * 64), 256, 0, stream>>>(m, (long)N * 64, 0.f);
    scatter_kernel<<<ew_blocks((long)E * 64), 256, 0, stream>>>(src, dst, hq, pre2, m, E);
    gemm(stream, m, 64, it[l].p0_w, 64, it[l].p0_b, tn, 64, N, 64, 64, FLAG_SSP);     // t = ssp(m@p0+b)
    gemm(stream, tn, 64, it[l].p1_w, H_, it[l].p1_b, h, H_, N, H_, 64, FLAG_ACC);     // h += t@p1+b

    ea_w = 64;
  }

  // output head: tn = ssp(h @ out0 + b)  [N,32]; then reduce over batch
  gemm(stream, h, H_, out0_w, 32, out0_b, tn, 32, N, 32, H_, FLAG_SSP);
  fill_kernel<<<1, 256, 0, stream>>>((float*)d_out, (long)out_size, 0.f);
  out_reduce_kernel<<<(N + 255) / 256, 256, 0, stream>>>(tn, out1_w, out1_b, batch, (float*)d_out, N);
}
